// DifferentiableDiagAstar_76501957476923
// MI455X (gfx1250) — matrix operand order repaired and checked by reading, not compiled, not run on hardware
//
#include <hip/hip_runtime.h>
#include <hip/hip_bf16.h>
#include <stdint.h>

// ---------------------------------------------------------------------------
// Differentiable diagonal A* (forward values only).
// B=32 items, 32x32 grid, <=972 sequential steps.
// Design: 1 wave32 per batch item (single-wave blocks, zero barriers needed),
// state in LDS + lane bitmasks, incremental wave-cooperative argmin.
// CDNA5 paths: v_wmma_f32_16x16x32_f16 for the goal-coordinate einsum,
// global_load_async_to_lds_b128 (+ s_wait_asynccnt) for global->LDS staging.
// ---------------------------------------------------------------------------

typedef __attribute__((ext_vector_type(16))) _Float16 v16h;
typedef __attribute__((ext_vector_type(8)))  float    v8f;

#define NCELL  1024
#define NSTEPS 972
#define SQRT2F 1.41421356237309515f   // rounds to float32(sqrt(2))

__device__ __forceinline__ float hfun(int r, int c, float gi, float gj) {
  // diag heuristic: min(dx,dy)*sqrt2 + |dx-dy|  (H_W = 1.0)
  float dx = fabsf((float)r - gi);
  float dy = fabsf((float)c - gj);
  return fminf(dx, dy) * SQRT2F + fabsf(dx - dy);
}

// f >= 0 always, so float bits are order-preserving; low 32 bits hold the
// flat index so uint64-min == lexicographic (f, index) min => reference's
// first-index tie-break for argmax(exp(-f/c)*open).
__device__ __forceinline__ unsigned long long packkey(float f, int ind) {
  return ((unsigned long long)__float_as_uint(f) << 32) | (unsigned)ind;
}

__device__ __forceinline__ unsigned long long wave_min64(unsigned long long k) {
#pragma unroll
  for (int m = 16; m >= 1; m >>= 1) {
    unsigned lo = __shfl_xor((unsigned)k, m, 32);
    unsigned hi = __shfl_xor((unsigned)(k >> 32), m, 32);
    unsigned long long o = ((unsigned long long)hi << 32) | lo;
    if (o < k) k = o;
  }
  return k;
}

// ---------------------------------------------------------------------------
// Kernel 0: goal coordinates via WMMA.
// Reference: goal_loc[b,k] = sum_ij loc[k,ij]*goal[b,ij]  (einsum = matmul).
// A = goal maps [16 x 1024] (f16, one-hot -> exact), B[k, n]: n=0 -> row(k),
// n=1 -> col(k). K chunk kk of 32 == grid row kk, so B col0 = kk, col1 = klocal.
// Products/sums <= 31 -> exact in f16 with f32 accumulate.
// ---------------------------------------------------------------------------
__global__ __launch_bounds__(32) void coords_wmma_kernel(
    const float* __restrict__ goal, float* __restrict__ coords) {
  const int lane  = threadIdx.x & 31;
  const int M     = lane & 15;              // A-matrix row within tile; C/D column N
  const int kbase = (lane < 16) ? 0 : 8;    // A layout: lanes16-31 start at K=8
  const int khi   = (lane < 16) ? 0 : 16;   // unused; B uses kbase, same K order as A

  for (int m0 = 0; m0 < 32; m0 += 16) {
    const float* gm = goal + (size_t)(m0 + M) * NCELL;
    v8f acc = {};
    for (int kk = 0; kk < 32; ++kk) {
      // A: 16-bit 16x32 layout — lanes0-15: V0..3 -> K0..7, V4..7 -> K16..23;
      //                          lanes16-31: V0..3 -> K8..15, V4..7 -> K24..31.
      v16h A;
#pragma unroll
      for (int hh = 0; hh < 16; ++hh) {
        int v  = hh >> 1;
        int kl = ((v < 4) ? (kbase + 2 * v) : (16 + kbase + 2 * (v - 4))) + (hh & 1);
        A[hh] = (_Float16)gm[kk * 32 + kl];
      }
      // B: 32x16, same K order as A (lanes0-15: K0..7,16..23; lanes16-31: K8..15,24..31); N = lane%16.
      v16h Bm;
#pragma unroll
      for (int hh = 0; hh < 16; ++hh) {
        int kl = (hh < 8) ? (kbase + hh) : (16 + kbase + hh - 8);
        float bv = (M == 0) ? (float)kk : ((M == 1) ? (float)kl : 0.0f);
        Bm[hh] = (_Float16)bv;
      }
      acc = __builtin_amdgcn_wmma_f32_16x16x32_f16(
          /*neg_a=*/false, A, /*neg_b=*/false, Bm,
          /*c_mod=*/(short)0, acc, /*reuse_a=*/false, /*reuse_b=*/false);
    }
    // C/D 16x16 f32 layout: lane L: N = L&15; VGPR k -> M = k (L<16) else 8+k.
#pragma unroll
    for (int k = 0; k < 8; ++k) {
      int bb = m0 + ((lane < 16) ? k : (8 + k));
      if (M == 0) coords[bb * 2 + 0] = acc[k];   // gi (row)
      if (M == 1) coords[bb * 2 + 1] = acc[k];   // gj (col)
    }
  }
}

// ---------------------------------------------------------------------------
// Kernel 1: A* core + backtrack. One wave per batch item.
// Layout: lane = column c; bit r of open/hist/obst masks = row r.
// g, parents in LDS (random access by flat index); per-column min-key cached
// in `colkey` register; only dirty columns (ci-1..ci+1) rescanned per step,
// cooperatively (lane = row).
// ---------------------------------------------------------------------------
__global__ __launch_bounds__(32) void astar_main_kernel(
    const float* __restrict__ start, const float* __restrict__ obst,
    const float* __restrict__ coords, float* __restrict__ out) {
  __shared__ float               g_lds[NCELL];
  __shared__ int                 par_lds[NCELL];
  __shared__ __align__(16) float stage[NCELL];
  __shared__ unsigned            pathw[32];

  const int lane = threadIdx.x & 31;
  const int b    = blockIdx.x;

  // --- async stage obstacle map into LDS (ASYNCcnt path, 8 x 512B) ---------
  unsigned           stage_addr = (unsigned)(uintptr_t)(&stage[0]);
  unsigned long long gaddr      = (unsigned long long)(uintptr_t)(obst + (size_t)b * NCELL);
#pragma unroll
  for (int i = 0; i < 8; ++i) {
    unsigned           la = stage_addr + (unsigned)(i * 512 + lane * 16);
    unsigned long long ga = gaddr + (unsigned long long)(i * 512 + lane * 16);
    asm volatile("global_load_async_to_lds_b128 %0, %1, off"
                 :: "v"(la), "v"(ga) : "memory");
  }

  // --- goal coords (from WMMA kernel) + init (overlaps async copy) ---------
  float gi = coords[b * 2 + 0];
  float gj = coords[b * 2 + 1];
  int goal_idx = (int)(gi + 0.5f) * 32 + (int)(gj + 0.5f);

  unsigned openBits = 0;   // g0 = 0, hist0 = 0, open = start map
#pragma unroll 4
  for (int r = 0; r < 32; ++r) {
    int cell = r * 32 + lane;
    g_lds[cell]   = 0.0f;
    par_lds[cell] = goal_idx;  // parents init = goal index (matches reference)
    openBits |= (start[(size_t)b * NCELL + cell] > 0.5f ? 1u : 0u) << r;
  }
  pathw[lane] = 0u;

  asm volatile("s_wait_asynccnt 0" ::: "memory");
  __syncthreads();  // single-wave: S_NOP barrier, also a compiler fence

  unsigned obstBits = 0;
#pragma unroll 4
  for (int r = 0; r < 32; ++r)
    obstBits |= (stage[r * 32 + lane] > 0.5f ? 1u : 0u) << r;

  unsigned histBits = 0;

  // initial per-column cached key (empty column -> (inf, index r=0,c=lane),
  // reproducing argmax-of-all-equal -> flat index 0 when nothing is open)
  unsigned long long colkey = packkey(__builtin_inff(), lane);
  for (int r = 0; r < 32; ++r) {
    if (openBits & (1u << r)) {
      float f = 0.5f * g_lds[r * 32 + lane] + 0.501f * hfun(r, lane, gi, gj);
      unsigned long long k = packkey(f, r * 32 + lane);
      if (k < colkey) colkey = k;
    }
  }

  // --- main sequential loop ------------------------------------------------
  for (int step = 0; step < NSTEPS; ++step) {
    unsigned long long best = wave_min64(colkey);      // global argmin(f, idx)
    int ind = (int)(best & 0xFFFFFFFFull);
    int ri = ind >> 5, ci = ind & 31;
    int solved = (ind == goal_idx);                    // dist >= 1e-8 case
    float g_ind = g_lds[ind];                          // broadcast LDS read

    if (lane == ci) {
      histBits |= 1u << ri;                            // hist = clip(hist+sel)
      if (!solved) openBits &= ~(1u << ri);            // open -= unsolved*sel
    }

    // sparse neighbor updates: conv3x3(sel, nf/cf) only touches ri+-1, ci+-1
    int dc = lane - ci;
    if (dc >= -1 && dc <= 1) {
#pragma unroll
      for (int dr = -1; dr <= 1; ++dr) {
        int nr = ri + dr;
        if (nr < 0 || nr > 31) continue;
        if (dr == 0 && dc == 0) continue;              // nf/cf center = 0
        unsigned nbit = 1u << nr;
        if (obstBits & nbit) {                         // nbr = conv*obst
          float cost = (dr != 0 && dc != 0) ? SQRT2F : 1.0f;  // cf weights
          float g2 = g_ind + cost;
          int n = nr * 32 + lane;
          bool ob = (openBits & nbit) != 0;            // open AFTER removal
          bool hb = (histBits & nbit) != 0;
          bool upd = ob ? (g_lds[n] > g2) : !hb;       // idx formula
          if (upd) {
            g_lds[n]   = g2;                           // g = g2*idx + g*(1-idx)
            par_lds[n] = ind;                          // parents <- ind
            openBits  |= nbit;                         // open += idx
          }
        }
      }
    }

    // once solved, state is provably stationary (goal stays argmin, updates
    // idempotent) -> per-item early exit == reference's global freeze
    if (solved) break;
    __syncthreads();

    // cooperative rescan of the <=3 dirty columns (lane = row)
#pragma unroll
    for (int d = -1; d <= 1; ++d) {
      int cd = ci + d;
      if (cd < 0 || cd > 31) continue;                 // uniform branch
      unsigned opencd = __shfl(openBits, cd, 32);      // column cd's open mask
      unsigned long long k = packkey(__builtin_inff(), cd);
      if (opencd & (1u << lane)) {
        float f = 0.5f * g_lds[lane * 32 + cd] + 0.501f * hfun(lane, cd, gi, gj);
        k = packkey(f, lane * 32 + cd);
      }
      k = wave_min64(k);
      if (lane == cd) colkey = k;
    }
  }

  __syncthreads();

  // --- backtrack: path = goal one-hot; loc = parents[goal]; chase pointers.
  // Early exit on returning to goal == reference's idempotent N_STEPS loop.
  if (lane == 0) {
    pathw[goal_idx & 31] |= 1u << (goal_idx >> 5);
    int loc = par_lds[goal_idx];
    for (int t = 0; t < NSTEPS; ++t) {
      pathw[loc & 31] |= 1u << (loc >> 5);
      if (loc == goal_idx) break;
      loc = par_lds[loc];
    }
  }
  __syncthreads();

  // --- outputs: hist | path | g, each [B,1,32,32] f32, concatenated --------
  float* outHist = out;
  float* outPath = out + 32 * NCELL;
  float* outG    = out + 2 * 32 * NCELL;
  unsigned pw = pathw[lane];
#pragma unroll 4
  for (int r = 0; r < 32; ++r) {
    int cell = r * 32 + lane;
    outHist[(size_t)b * NCELL + cell] = ((histBits >> r) & 1u) ? 1.0f : 0.0f;
    outPath[(size_t)b * NCELL + cell] = ((pw >> r) & 1u) ? 1.0f : 0.0f;
    outG[(size_t)b * NCELL + cell]    = g_lds[cell];
  }
}

extern "C" void kernel_launch(void* const* d_in, const int* in_sizes, int n_in,
                              void* d_out, int out_size, void* d_ws, size_t ws_size,
                              hipStream_t stream) {
  (void)in_sizes; (void)n_in; (void)out_size; (void)ws_size;
  // setup_inputs order: cost_maps (unused in 'default' mode), start, goal, obst
  const float* start = (const float*)d_in[1];
  const float* goal  = (const float*)d_in[2];
  const float* obst  = (const float*)d_in[3];
  float* coords = (float*)d_ws;  // 32 x {gi, gj}

  coords_wmma_kernel<<<1, 32, 0, stream>>>(goal, coords);
  astar_main_kernel<<<32, 32, 0, stream>>>(start, obst, coords, (float*)d_out);
}
